// CoherenceAttention_24713241821750
// MI455X (gfx1250) — compile-verified, matrix operand order checked by reading, not run on hardware
//
#include <hip/hip_runtime.h>
#include <hip/hip_bf16.h>

#define B_   8
#define T_   4096
#define H_   1024
#define S_   64
#define NH_  16
#define DH_  64

typedef __attribute__((ext_vector_type(16))) __bf16 v16bf;
typedef __attribute__((ext_vector_type(8)))  __bf16 bf16x8;
typedef __attribute__((ext_vector_type(8)))  float  v8f;

#define SHUF16(lo, hi) __builtin_shufflevector(lo, hi, 0,1,2,3,4,5,6,7,8,9,10,11,12,13,14,15)

// ---------------------------------------------------------------------------
// f32 -> bf16 weight conversion (strided source, packed dest)
// ---------------------------------------------------------------------------
__global__ void cvt_bf16_kernel(const float* __restrict__ src, __bf16* __restrict__ dst,
                                int rows, int cols, int srcLd, int srcOff) {
    int i = blockIdx.x * blockDim.x + threadIdx.x;
    if (i >= rows * cols) return;
    int r = i / cols, c = i - r * cols;
    dst[(size_t)r * cols + c] = (__bf16)src[(size_t)r * srcLd + srcOff + c];
}

// ---------------------------------------------------------------------------
// Sentence means: one block per (b,s); 256 threads x 4 channels
// ---------------------------------------------------------------------------
__global__ void sent_mean_kernel(const float* __restrict__ h, const int* __restrict__ bounds,
                                 float* __restrict__ srF, __bf16* __restrict__ srB) {
    int bs = blockIdx.x;                       // b*S + s
    int b  = bs / S_;
    int start = bounds[bs * 2 + 0];
    int end   = bounds[bs * 2 + 1];
    int lo = start < 0 ? 0 : start;
    int hi = end > T_ ? T_ : end;
    int cnt = hi - lo; if (cnt < 1) cnt = 1;   // counts = clip(sum, 1.0)
    float inv = 1.0f / (float)cnt;
    int ch = threadIdx.x * 4;
    float a0 = 0.f, a1 = 0.f, a2 = 0.f, a3 = 0.f;
    for (int t = lo; t < hi; ++t) {
        const float4 hv = *(const float4*)(h + ((size_t)b * T_ + t) * H_ + ch);
        a0 += hv.x; a1 += hv.y; a2 += hv.z; a3 += hv.w;
    }
    size_t o = (size_t)bs * H_ + ch;
    float r0 = a0 * inv, r1 = a1 * inv, r2 = a2 * inv, r3 = a3 * inv;
    srF[o + 0] = r0; srF[o + 1] = r1; srF[o + 2] = r2; srF[o + 3] = r3;
    srB[o + 0] = (__bf16)r0; srB[o + 1] = (__bf16)r1;
    srB[o + 2] = (__bf16)r2; srB[o + 3] = (__bf16)r3;
}

// ---------------------------------------------------------------------------
// Per-token sentence id (first member, argmax semantics) + covered flag
// ---------------------------------------------------------------------------
__global__ void sid_kernel(const int* __restrict__ bounds,
                           int* __restrict__ sid, float* __restrict__ cov) {
    int row = blockIdx.x * blockDim.x + threadIdx.x;   // b*T + t
    if (row >= B_ * T_) return;
    int b = row / T_, t = row - b * T_;
    int found = 0, sv = 0;
    for (int s = 0; s < S_; ++s) {
        int st = bounds[(b * S_ + s) * 2 + 0];
        int en = bounds[(b * S_ + s) * 2 + 1];
        if (!found && t >= st && t < en) { sv = s; found = 1; }
    }
    sid[row] = sv;
    cov[row] = found ? 1.0f : 0.0f;
}

// ---------------------------------------------------------------------------
// Generic WMMA GEMM:  C[M,N] = A[M,K] * B[N,K]^T (+bias[n]); bf16 in, f32 acc.
// One 16x16 C tile per wave; 8 waves per block. M,N multiples of 16; K mult 32.
// ---------------------------------------------------------------------------
__global__ void wmma_gemm_bf16(const __bf16* __restrict__ A, const __bf16* __restrict__ Bm,
                               const float* __restrict__ bias,
                               float* __restrict__ Cf, __bf16* __restrict__ Cb,
                               int M, int N, int K) {
    int tid = threadIdx.x;
    int wave = tid >> 5, lane = tid & 31;
    int halfSel = lane >> 4, lnib = lane & 15;
    int tilesN = N >> 4;
    int tileId = blockIdx.x * 8 + wave;
    int tm = (tileId / tilesN) << 4;
    int tn = (tileId - (tileId / tilesN) * tilesN) << 4;
    if (tm >= M) return;                                   // wave-uniform guard

    v8f acc = (v8f)0.0f;
    const __bf16* aRow = A  + (size_t)(tm + lnib) * K + halfSel * 8;
    const __bf16* bRow = Bm + (size_t)(tn + lnib) * K + halfSel * 8;
    for (int k0 = 0; k0 < K; k0 += 32) {
        bf16x8 al = *(const bf16x8*)(aRow + k0);
        bf16x8 ah = *(const bf16x8*)(aRow + k0 + 16);
        bf16x8 bl = *(const bf16x8*)(bRow + k0);
        bf16x8 bh = *(const bf16x8*)(bRow + k0 + 16);
        v16bf af = SHUF16(al, ah);
        v16bf bf = SHUF16(bl, bh);
        acc = __builtin_amdgcn_wmma_f32_16x16x32_bf16(false, af, false, bf,
                                                      (short)0, acc, false, false);
    }
#pragma unroll
    for (int v = 0; v < 8; ++v) {
        int m = tm + v + 8 * halfSel;
        int n = tn + lnib;
        float val = acc[v] + (bias ? bias[n] : 0.0f);
        if (Cf) Cf[(size_t)m * N + n] = val;
        if (Cb) Cb[(size_t)m * N + n] = (__bf16)val;
    }
}

// ---------------------------------------------------------------------------
// Attention over S=64 per (b,nh). f32 scalar (67 MFLOP total), ctx out as bf16.
// ---------------------------------------------------------------------------
__global__ void attn_kernel(const float* __restrict__ qkv, __bf16* __restrict__ ctxB) {
    __shared__ float qs[64][64], ks[64][64], vs[64][64];
    int bh = blockIdx.x;
    int b = bh / NH_, nh = bh - b * NH_;
    int tid = threadIdx.x;                     // 64 threads
    for (int i = tid; i < 64 * 64; i += 64) {
        int s = i >> 6, d = i & 63;
        size_t base = (size_t)(b * S_ + s) * (3 * H_) + nh * DH_ + d;
        qs[s][d] = qkv[base];
        ks[s][d] = qkv[base + H_];
        vs[s][d] = qkv[base + 2 * H_];
    }
    __syncthreads();
    int qi = tid;
    float sc[64];
    float mx = -1e30f;
    for (int kj = 0; kj < 64; ++kj) {
        float dot = 0.f;
#pragma unroll 8
        for (int d = 0; d < 64; ++d) dot += qs[qi][d] * ks[kj][d];
        dot *= 0.125f;                          // 1/sqrt(64)
        sc[kj] = dot;
        mx = fmaxf(mx, dot);
    }
    float den = 0.f;
    for (int kj = 0; kj < 64; ++kj) { float e = __expf(sc[kj] - mx); sc[kj] = e; den += e; }
    float rcp = 1.0f / den;
    for (int d = 0; d < 64; ++d) {
        float a = 0.f;
#pragma unroll 8
        for (int kj = 0; kj < 64; ++kj) a += sc[kj] * vs[kj][d];
        ctxB[(size_t)(b * S_ + qi) * H_ + nh * DH_ + d] = (__bf16)(a * rcp);
    }
}

// ---------------------------------------------------------------------------
// Fused: score[row] = sum_n relu( h[row]·W1a[n] + attPart[b,sid,n] + b1[n] )*w2[n] + b2
//        out[row,:] = h[row,:] * (1 + covered*score)
// Block: 64 rows, 8 waves = 4 m-tiles x 2 n-groups. A panel (64x1024 bf16) staged
// once f32->bf16; each 64-wide N chunk of W1a staged via ASYNC global->LDS b128.
// ---------------------------------------------------------------------------
#define MBLK 64
#define LDT  1032            // LDS row stride (bf16 elems): 516 dwords -> conflict-free frags
__global__ void fused_mlp_score_kernel(const float* __restrict__ h,
                                       const __bf16* __restrict__ w1a,
                                       const float* __restrict__ b1,
                                       const float* __restrict__ w2,
                                       const float* __restrict__ b2,
                                       const float* __restrict__ attPart,
                                       const int* __restrict__ sid,
                                       const float* __restrict__ cov,
                                       float* __restrict__ out) {
    extern __shared__ __align__(16) char smem[];
    __bf16* As     = (__bf16*)smem;                                   // [64][LDT]
    __bf16* Bs     = (__bf16*)(smem + (size_t)MBLK * LDT * 2);        // [64][LDT]
    float*  scoreS = (float*)(smem + (size_t)2 * MBLK * LDT * 2);     // [64]

    int tid = threadIdx.x;
    int rowBase = blockIdx.x * MBLK;

    if (tid < MBLK) scoreS[tid] = 0.f;

    // Stage A panel: 64 rows of h, f32 -> bf16 in LDS
    for (int i = tid; i < MBLK * (H_ / 4); i += 256) {
        int r  = i >> 8;
        int c4 = (i & 255) << 2;
        const float4 hv = *(const float4*)(h + (size_t)(rowBase + r) * H_ + c4);
        __bf16* dst = As + r * LDT + c4;
        dst[0] = (__bf16)hv.x; dst[1] = (__bf16)hv.y;
        dst[2] = (__bf16)hv.z; dst[3] = (__bf16)hv.w;
    }
    __syncthreads();

    int wave = tid >> 5, lane = tid & 31;
    int halfSel = lane >> 4, lnib = lane & 15;
    int mt = wave & 3;                       // m-tile 0..3
    int ng = wave >> 2;                      // n-group 0..1 (handles nt = ng, ng+2)
    int waveRow = mt * 16;

    // Per-row metadata (uniform across the 16 lanes of each half)
    int attBase[8];
#pragma unroll
    for (int v = 0; v < 8; ++v) {
        int rG = rowBase + waveRow + v + 8 * halfSel;
        int bI = rG / T_;
        attBase[v] = (bI * S_ + sid[rG]) * H_;
    }

    float sacc[2][8];
#pragma unroll
    for (int u = 0; u < 2; ++u)
#pragma unroll
        for (int v = 0; v < 8; ++v) sacc[u][v] = 0.f;

    const __bf16* aRowL = As + (waveRow + lnib) * LDT + halfSel * 8;
    unsigned bsBase = (unsigned)(size_t)Bs;           // LDS byte offset (addr[31:0])

    for (int nc = 0; nc < H_; nc += 64) {
        // Async-stage B chunk: w1a rows [nc, nc+64) x 1024 K -> LDS (128 KB)
        // 8192 x 16B chunks, 32 per thread, tracked by ASYNCcnt.
        for (int i = tid; i < MBLK * 128; i += 256) {
            int r  = i >> 7;                  // row in chunk
            int ck = i & 127;                 // 16B chunk within row
            unsigned long long ga =
                (unsigned long long)(size_t)(w1a + (size_t)(nc + r) * H_ + ck * 8);
            unsigned la = bsBase + (unsigned)(r * (LDT * 2) + ck * 16);
            asm volatile("global_load_async_to_lds_b128 %0, %1, off"
                         :: "v"(la), "v"(ga) : "memory");
        }
        asm volatile("s_wait_asynccnt 0x0" ::: "memory");
        __syncthreads();                      // B chunk visible to all waves

        v8f acc[2];
#pragma unroll
        for (int u = 0; u < 2; ++u) acc[u] = (v8f)0.0f;

        for (int k0 = 0; k0 < H_; k0 += 32) {
            bf16x8 al = *(const bf16x8*)(aRowL + k0);
            bf16x8 ah = *(const bf16x8*)(aRowL + k0 + 16);
            v16bf af = SHUF16(al, ah);
#pragma unroll
            for (int u = 0; u < 2; ++u) {
                int nt = ng + u * 2;          // 0..3
                const __bf16* bRowL = Bs + (nt * 16 + lnib) * LDT + halfSel * 8 + k0;
                bf16x8 bl = *(const bf16x8*)(bRowL);
                bf16x8 bh = *(const bf16x8*)(bRowL + 16);
                v16bf bf = SHUF16(bl, bh);
                acc[u] = __builtin_amdgcn_wmma_f32_16x16x32_bf16(false, af, false, bf,
                                                                 (short)0, acc[u], false, false);
            }
        }

        // Epilogue: bias + att gather + relu, dot with w2
#pragma unroll
        for (int u = 0; u < 2; ++u) {
            int nt = ng + u * 2;
            int n  = nc + nt * 16 + lnib;
            float bn  = b1[n];
            float w2n = w2[n];
#pragma unroll
            for (int v = 0; v < 8; ++v) {
                float val = acc[u][v] + bn + attPart[attBase[v] + n];
                val = fmaxf(val, 0.f);
                sacc[u][v] += val * w2n;
            }
        }
        __syncthreads();                      // done reading Bs before next overwrite
    }

    // Reduce each row's partial across its 16-lane group; combine 2 n-group waves
#pragma unroll
    for (int u = 0; u < 2; ++u) {
#pragma unroll
        for (int v = 0; v < 8; ++v) {
            float s = sacc[u][v];
            s += __shfl_xor(s, 8, 16);
            s += __shfl_xor(s, 4, 16);
            s += __shfl_xor(s, 2, 16);
            s += __shfl_xor(s, 1, 16);
            if (lnib == 0)
                atomicAdd(&scoreS[waveRow + v + 8 * halfSel], s);
        }
    }
    __syncthreads();

    // out = h * (1 + covered * (score + b2))
    float b2v = b2[0];
    for (int i = tid; i < MBLK * (H_ / 4); i += 256) {
        int r  = i >> 8;
        int c4 = (i & 255) << 2;
        size_t o = (size_t)(rowBase + r) * H_ + c4;
        float4 hv = *(const float4*)(h + o);
        float sc = 1.0f + (scoreS[r] + b2v) * cov[rowBase + r];
        float4 ov; ov.x = hv.x * sc; ov.y = hv.y * sc; ov.z = hv.z * sc; ov.w = hv.w * sc;
        *(float4*)(out + o) = ov;
    }
}

// ---------------------------------------------------------------------------
extern "C" void kernel_launch(void* const* d_in, const int* in_sizes, int n_in,
                              void* d_out, int out_size, void* d_ws, size_t ws_size,
                              hipStream_t stream) {
    const float* h      = (const float*)d_in[0];
    const int*   bounds = (const int*)  d_in[1];
    const float* ipw    = (const float*)d_in[2];
    const float* ipb    = (const float*)d_in[3];
    const float* ow     = (const float*)d_in[4];
    const float* ob     = (const float*)d_in[5];
    const float* w1     = (const float*)d_in[6];
    const float* b1     = (const float*)d_in[7];
    const float* w2     = (const float*)d_in[8];
    const float* b2     = (const float*)d_in[9];
    float* out = (float*)d_out;

    // Workspace carve-up (256B aligned), ~25.25 MiB total
    char* ws = (char*)d_ws;
    size_t off = 0;
    auto carve = [&](size_t bytes) { char* p = ws + off; off = (off + bytes + 255) & ~(size_t)255; return p; };
    float*  srF   = (float*) carve((size_t)B_ * S_ * H_ * 4);          // sentence means f32
    __bf16* srB   = (__bf16*)carve((size_t)B_ * S_ * H_ * 2);          // sentence means bf16
    float*  qkvF  = (float*) carve((size_t)B_ * S_ * 3 * H_ * 4);      // qkv f32
    __bf16* ctxB  = (__bf16*)carve((size_t)B_ * S_ * H_ * 2);          // attention ctx bf16
    __bf16* attB  = (__bf16*)carve((size_t)B_ * S_ * H_ * 2);          // attended bf16
    float*  attP  = (float*) carve((size_t)B_ * S_ * H_ * 4);          // attended @ W1b^T f32
    int*    sidA  = (int*)   carve((size_t)B_ * T_ * 4);
    float*  covA  = (float*) carve((size_t)B_ * T_ * 4);
    __bf16* ipwB  = (__bf16*)carve((size_t)3 * H_ * H_ * 2);
    __bf16* owB   = (__bf16*)carve((size_t)H_ * H_ * 2);
    __bf16* w1aB  = (__bf16*)carve((size_t)H_ * H_ * 2);
    __bf16* w1bB  = (__bf16*)carve((size_t)H_ * H_ * 2);

    // Weight conversions f32 -> bf16
    {
        int n;
        n = 3 * H_ * H_;
        cvt_bf16_kernel<<<(n + 255) / 256, 256, 0, stream>>>(ipw, ipwB, 3 * H_, H_, H_, 0);
        n = H_ * H_;
        cvt_bf16_kernel<<<(n + 255) / 256, 256, 0, stream>>>(ow,  owB,  H_, H_, H_, 0);
        cvt_bf16_kernel<<<(n + 255) / 256, 256, 0, stream>>>(w1,  w1aB, H_, H_, 2 * H_, 0);
        cvt_bf16_kernel<<<(n + 255) / 256, 256, 0, stream>>>(w1,  w1bB, H_, H_, 2 * H_, H_);
    }

    // Sentence means + per-token sid/covered
    sent_mean_kernel<<<B_ * S_, 256, 0, stream>>>(h, bounds, srF, srB);
    sid_kernel<<<(B_ * T_ + 255) / 256, 256, 0, stream>>>(bounds, sidA, covA);

    // qkv = sent_repr @ in_proj_w^T + in_proj_b        (512 x 3072 x 1024)
    {
        int M = B_ * S_, N = 3 * H_, K = H_;
        int blocks = (M / 16) * (N / 16) / 8;
        wmma_gemm_bf16<<<blocks, 256, 0, stream>>>(srB, ipwB, ipb, qkvF, nullptr, M, N, K);
    }

    // Attention (per b,nh), ctx -> bf16
    attn_kernel<<<B_ * NH_, 64, 0, stream>>>(qkvF, ctxB);

    // attended = ctx @ out_w^T + out_b  -> bf16        (512 x 1024 x 1024)
    {
        int M = B_ * S_, N = H_, K = H_;
        int blocks = (M / 16) * (N / 16) / 8;
        wmma_gemm_bf16<<<blocks, 256, 0, stream>>>(ctxB, owB, ob, nullptr, attB, M, N, K);
    }

    // attPart = attended @ W1b^T  -> f32               (512 x 1024 x 1024)
    {
        int M = B_ * S_, N = H_, K = H_;
        int blocks = (M / 16) * (N / 16) / 8;
        wmma_gemm_bf16<<<blocks, 256, 0, stream>>>(attB, w1bB, nullptr, attP, nullptr, M, N, K);
    }

    // Fused big GEMM + relu + w2 reduction + residual scale (32768 x 1024 x 1024)
    {
        int blocks = (B_ * T_) / MBLK;                         // 512
        size_t shmem = (size_t)2 * MBLK * LDT * sizeof(__bf16) // A + B panels (264 KB)
                     + MBLK * sizeof(float);                   // scores
        fused_mlp_score_kernel<<<blocks, 256, shmem, stream>>>(
            h, w1aB, b1, w2, b2, attP, sidA, covA, out);
    }
}
